// CAM_50689204027802
// MI455X (gfx1250) — matrix operand order repaired and checked by reading, not compiled, not run on hardware
//
#include <hip/hip_runtime.h>
#include <hip/hip_bf16.h>

// ---------------------------------------------------------------------------
// Channel Attention Module (SAGAN CAM) for MI455X / gfx1250 (wave32, WMMA).
//   x: [B=16, C=512, H=64, W=64] f32, gamma: [1] f32
//   energy = Q Q^T (Q = x reshaped [B,C,N], N=4096)
//   attention = softmax(rowmax - energy) == softmax(-energy)   (row-wise)
//   out = attention @ Q ;  y = gamma*out + x
// FP16 WMMA (v_wmma_f32_16x16x32_f16) with fp32 accumulation.
// Staging uses 128-bit global loads and packed 64/128-bit LDS stores.
// ---------------------------------------------------------------------------

typedef __attribute__((ext_vector_type(16))) _Float16 v16h;
typedef __attribute__((ext_vector_type(8)))  _Float16 v8h;
typedef __attribute__((ext_vector_type(4)))  _Float16 v4h;
typedef __attribute__((ext_vector_type(8)))  float    v8f;
typedef __attribute__((ext_vector_type(4)))  float    v4f;

namespace cam {
constexpr int B  = 16;
constexpr int C  = 512;
constexpr int N  = 4096;          // H*W
constexpr int TK = 32;            // WMMA K per step (f16)
constexpr int LDK = TK + 8;       // LDS row stride in halves: 80B, 8B/16B-aligned, conflict-free
}

__device__ inline v16h cam_combine(v8h lo, v8h hi) {
  v16h r;
#pragma unroll
  for (int i = 0; i < 8; ++i) { r[i] = lo[i]; r[i + 8] = hi[i]; }
  return r;
}

__device__ inline v4h cam_cvt4(v4f v) {
  v4h h;
#pragma unroll
  for (int i = 0; i < 4; ++i) h[i] = (_Float16)v[i];
  return h;
}

// A-matrix fragment (16x32 f16): lane L<16 -> row L,    K = {0..7, 16..23}
//                                lane L>=16 -> row L-16, K = {8..15, 24..31}
__device__ inline v16h cam_load_afrag(const _Float16* lds, int rowBase, int lane) {
  const int row  = rowBase + (lane & 15);
  const int kOff = (lane < 16) ? 0 : 8;
  const _Float16* p = lds + row * cam::LDK + kOff;
  v8h lo = *(const v8h*)(p);
  v8h hi = *(const v8h*)(p + 16);
  return cam_combine(lo, hi);
}

// B-matrix fragment (32x16 f16): lane L -> column L%16, K = {0..7, 16..23} (L<16) / {8..15, 24..31} (L>=16)
// lds stored column-major-for-B: lds[col * LDK + k], k contiguous.
__device__ inline v16h cam_load_bfrag(const _Float16* lds, int colBase, int lane) {
  const int col  = colBase + (lane & 15);
  const int kOff = (lane < 16) ? 0 : 8;
  const _Float16* p = lds + col * cam::LDK + kOff;
  v8h lo = *(const v8h*)(p);
  v8h hi = *(const v8h*)(p + 16);
  return cam_combine(lo, hi);
}

// ---------------------------------------------------------------------------
// Kernel 1: energy[b, i, j] = sum_k Q[b,i,k] * Q[b,j,k]
// Grid: (C/64, C/64, B), block 128 (4 waves). 64x64 tile, K-loop 4096/32.
// ---------------------------------------------------------------------------
__global__ __launch_bounds__(128) void cam_energy_kernel(
    const float* __restrict__ x, float* __restrict__ energy) {
  using namespace cam;
  __shared__ alignas(16) _Float16 ldsA[64 * LDK];
  __shared__ alignas(16) _Float16 ldsB[64 * LDK];

  const int tid  = threadIdx.x;
  const int lane = tid & 31;
  const int wave = tid >> 5;
  const int j0 = blockIdx.x * 64;
  const int i0 = blockIdx.y * 64;
  const int b  = blockIdx.z;

  const float* xA = x + ((size_t)b * C + i0) * N;   // rows i (A operand)
  const float* xB = x + ((size_t)b * C + j0) * N;   // rows j (B operand, K-contiguous)

  v8f acc[4] = {};

  for (int kb = 0; kb < N; kb += TK) {
    // Stage 64x32 fp32 -> f16 into LDS: 512 float4 chunks, 4 per thread,
    // global_load_b128 + packed ds_store_b64.
#pragma unroll
    for (int e = 0; e < 4; ++e) {
      const int f4  = tid + e * 128;
      const int row = f4 >> 3;           // 8 float4 per 32-wide row
      const int c4  = (f4 & 7) * 4;
      v4f a4 = *(const v4f*)&xA[(size_t)row * N + kb + c4];
      v4f b4 = *(const v4f*)&xB[(size_t)row * N + kb + c4];
      *(v4h*)&ldsA[row * LDK + c4] = cam_cvt4(a4);
      *(v4h*)&ldsB[row * LDK + c4] = cam_cvt4(b4);
    }
    if (kb + TK < N) {
      // Prefetch next K-chunk (global_prefetch_b8); one line hint per thread.
      const int row = tid >> 1;
      const int col = (tid & 1) * 16;
      __builtin_prefetch((const void*)&xA[(size_t)row * N + kb + TK + col], 0, 1);
      __builtin_prefetch((const void*)&xB[(size_t)row * N + kb + TK + col], 0, 1);
    }
    __syncthreads();

    const int rBase = wave * 16;
    v16h a = cam_load_afrag(ldsA, rBase, lane);
#pragma unroll
    for (int c = 0; c < 4; ++c) {
      v16h bfr = cam_load_bfrag(ldsB, c * 16, lane);
      acc[c] = __builtin_amdgcn_wmma_f32_16x16x32_f16(
          false, a, false, bfr, (short)0, acc[c], false, false);
    }
    __syncthreads();
  }

  // C/D layout: VGPR r -> M = r + (lane<16 ? 0 : 8), N = lane%16.
  const int hi8  = (lane < 16) ? 0 : 8;
  const int colL = lane & 15;
  float* eT = energy + ((size_t)b * C + i0) * C + j0;
#pragma unroll
  for (int c = 0; c < 4; ++c) {
#pragma unroll
    for (int r = 0; r < 8; ++r) {
      eT[(size_t)(wave * 16 + r + hi8) * C + c * 16 + colL] = acc[c][r];
    }
  }
}

// ---------------------------------------------------------------------------
// Kernel 2: attention = softmax(-energy) per row, stored f16.
// One wave per row of 512; block 256 = 8 rows; grid = B*C/8 = 1024.
// ---------------------------------------------------------------------------
__global__ __launch_bounds__(256) void cam_softmax_kernel(
    const float* __restrict__ energy, _Float16* __restrict__ att) {
  using namespace cam;
  const int lane = threadIdx.x & 31;
  const int row  = blockIdx.x * 8 + (threadIdx.x >> 5);
  const float* e = energy + (size_t)row * C;

  float v[16];
#pragma unroll
  for (int t = 0; t < 4; ++t) {
    v4f tmp = *(const v4f*)&e[lane * 4 + t * 128];
#pragma unroll
    for (int i = 0; i < 4; ++i) v[t * 4 + i] = tmp[i];
  }

  float m = v[0];
#pragma unroll
  for (int t = 1; t < 16; ++t) m = fminf(m, v[t]);
#pragma unroll
  for (int off = 16; off > 0; off >>= 1) m = fminf(m, __shfl_xor(m, off, 32));

  float w[16];
  float s = 0.0f;
#pragma unroll
  for (int t = 0; t < 16; ++t) { w[t] = __expf(m - v[t]); s += w[t]; }
#pragma unroll
  for (int off = 16; off > 0; off >>= 1) s += __shfl_xor(s, off, 32);

  const float inv = 1.0f / s;
  _Float16* a = att + (size_t)row * C;
#pragma unroll
  for (int t = 0; t < 4; ++t) {
    v4h h;
#pragma unroll
    for (int i = 0; i < 4; ++i) h[i] = (_Float16)(w[t * 4 + i] * inv);
    *(v4h*)&a[lane * 4 + t * 128] = h;
  }
}

// ---------------------------------------------------------------------------
// Kernel 3: out[b,i,n] = sum_j att[b,i,j] * Q[b,j,n];  y = gamma*out + x
// Grid: (N/64, C/64, B), block 128. K = 512 in chunks of 32.
// B tile is transposed into LDS [n][k] so B fragments are K-contiguous.
// ---------------------------------------------------------------------------
__global__ __launch_bounds__(128) void cam_out_kernel(
    const float* __restrict__ x, const _Float16* __restrict__ att,
    const float* __restrict__ gamma, float* __restrict__ out) {
  using namespace cam;
  __shared__ alignas(16) _Float16 ldsA [64 * LDK];   // attention rows, K-contiguous
  __shared__ alignas(16) _Float16 ldsBT[64 * LDK];   // Q tile transposed: [n][k]

  const int tid  = threadIdx.x;
  const int lane = tid & 31;
  const int wave = tid >> 5;
  const int n0 = blockIdx.x * 64;
  const int i0 = blockIdx.y * 64;
  const int b  = blockIdx.z;

  const _Float16* aG = att + ((size_t)b * C + i0) * C;
  const float*    xG = x   + (size_t)b * C * N;

  v8f acc[4] = {};

  for (int kb = 0; kb < C; kb += TK) {
    // A: 64 rows x 32 k f16 -> 256 v8h chunks, 2 per thread (b128 load/store).
#pragma unroll
    for (int e = 0; e < 2; ++e) {
      const int c8  = tid + e * 128;
      const int ar  = c8 >> 2;           // 4 v8h per 32-wide row
      const int ak  = (c8 & 3) * 8;
      v8h av = *(const v8h*)&aG[(size_t)ar * C + kb + ak];
      *(v8h*)&ldsA[ar * LDK + ak] = av;
    }
    // B: 32 k-rows x 64 n fp32, float4 along n (b128 load), transpose on store.
#pragma unroll
    for (int e = 0; e < 4; ++e) {
      const int f4 = tid + e * 128;
      const int bk = f4 >> 4;            // 16 float4 per 64-wide n row
      const int n4 = (f4 & 15) * 4;
      v4f bv = *(const v4f*)&xG[(size_t)(kb + bk) * N + n0 + n4];
#pragma unroll
      for (int i = 0; i < 4; ++i) ldsBT[(n4 + i) * LDK + bk] = (_Float16)bv[i];
    }
    if (kb + TK < C) {
      const int bk = tid >> 2;
      const int bn = (tid & 3) * 16;
      __builtin_prefetch((const void*)&xG[(size_t)(kb + TK + bk) * N + n0 + bn], 0, 1);
    }
    __syncthreads();

    const int rBase = wave * 16;
    v16h a = cam_load_afrag(ldsA, rBase, lane);
#pragma unroll
    for (int c = 0; c < 4; ++c) {
      v16h bfr = cam_load_bfrag(ldsBT, c * 16, lane);
      acc[c] = __builtin_amdgcn_wmma_f32_16x16x32_f16(
          false, a, false, bfr, (short)0, acc[c], false, false);
    }
    __syncthreads();
  }

  const float g  = *gamma;
  const int hi8  = (lane < 16) ? 0 : 8;
  const int colL = lane & 15;
  const size_t base = ((size_t)b * C + i0) * N + n0;
#pragma unroll
  for (int c = 0; c < 4; ++c) {
#pragma unroll
    for (int r = 0; r < 8; ++r) {
      const size_t idx = base + (size_t)(wave * 16 + r + hi8) * N + c * 16 + colL;
      out[idx] = g * acc[c][r] + x[idx];
    }
  }
}

// ---------------------------------------------------------------------------
extern "C" void kernel_launch(void* const* d_in, const int* in_sizes, int n_in,
                              void* d_out, int out_size, void* d_ws, size_t ws_size,
                              hipStream_t stream) {
  using namespace cam;
  const float* x     = (const float*)d_in[0];
  const float* gamma = (const float*)d_in[1];
  float*       out   = (float*)d_out;

  // Workspace: energy f32 [B,C,C] (16 MB) + attention f16 [B,C,C] (8 MB).
  float*     energy = (float*)d_ws;
  _Float16*  att    = (_Float16*)((char*)d_ws + (size_t)B * C * C * sizeof(float));

  dim3 g1(C / 64, C / 64, B);
  cam_energy_kernel<<<g1, 128, 0, stream>>>(x, energy);

  cam_softmax_kernel<<<(B * C) / 8, 256, 0, stream>>>(energy, att);

  dim3 g3(N / 64, C / 64, B);
  cam_out_kernel<<<g3, 128, 0, stream>>>(x, att, gamma, out);
}
